// CapacitanceMatrix_62302795596479
// MI455X (gfx1250) — matrix operand order repaired and checked by reading, not compiled, not run on hardware
//
#include <hip/hip_runtime.h>
#include <stdint.h>

// ---------------------------------------------------------------------------
// CapacitanceMatrix: C[b,i,j] = sigmoid(x@Wd^T+bd) on diag,
//                               -softplus(x@Wo^T+bo) off-diag (row-major order)
// One 65536x1024x256 GEMM via v_wmma_f32_16x16x32_bf16 with hi/lo bf16 split
// (3 WMMA products ~ f32 accuracy), K-chunk double-buffered LDS pipeline
// (20KB LDS -> multi-block/WGP occupancy, HBM streaming overlapped with WMMA),
// fused activation epilogue.
// ---------------------------------------------------------------------------

#define D_DIM 1024
#define ROWS_PER_BLOCK 64            // 4 M-tiles of 16
#define MT 4
#define KT_SIZE 32                   // K per wmma_f32_16x16x32_bf16
#define NKT (D_DIM / KT_SIZE)        // 32
#define CH_ROW_ELEMS (KT_SIZE + 8)   // 40 bf16 (pad 8 -> 20-dword row stride)
#define CH_ROW_BYTES (CH_ROW_ELEMS * 2)             // 80
#define CH_PLANE_BYTES (ROWS_PER_BLOCK * CH_ROW_BYTES)  // 5120
#define CH_BUF_BYTES (2 * CH_PLANE_BYTES)           // 10240 (hi plane + lo plane)
#define SMEM_BYTES (2 * CH_BUF_BYTES)               // 20480 (double buffer)
#define WS_BYTES_NEEDED (16u * NKT * 2u * 32u * 32u)  // 1 MiB of B-fragments

typedef __attribute__((ext_vector_type(16))) __bf16 bf16x16;
typedef __attribute__((ext_vector_type(8)))  float  f32x8;
typedef __attribute__((ext_vector_type(4)))  float  f32x4;   // native vec for NT loads

struct U32x8 { uint4 lo, hi; };

__device__ __forceinline__ bf16x16 make_frag(uint4 a, uint4 b) {
  U32x8 t; t.lo = a; t.hi = b;
  return __builtin_bit_cast(bf16x16, t);
}

__device__ __forceinline__ unsigned bf16_rne(float f) {
  unsigned u = __float_as_uint(f);
  return (u + 0x7fffu + ((u >> 16) & 1u)) >> 16;
}

__device__ __forceinline__ f32x8 zero8() {
  f32x8 z = {0.f, 0.f, 0.f, 0.f, 0.f, 0.f, 0.f, 0.f};
  return z;
}

struct HiLo { uint2 h, l; };
// Split 4 f32 -> packed bf16 hi (truncated) and lo (RNE of residual).
__device__ __forceinline__ HiLo split4(f32x4 v) {
  unsigned u0 = __float_as_uint(v.x), u1 = __float_as_uint(v.y);
  unsigned u2 = __float_as_uint(v.z), u3 = __float_as_uint(v.w);
  HiLo r;
  r.h.x = (u0 >> 16) | ((u1 >> 16) << 16);
  r.h.y = (u2 >> 16) | ((u3 >> 16) << 16);
  float l0 = v.x - __uint_as_float(u0 & 0xffff0000u);
  float l1 = v.y - __uint_as_float(u1 & 0xffff0000u);
  float l2 = v.z - __uint_as_float(u2 & 0xffff0000u);
  float l3 = v.w - __uint_as_float(u3 & 0xffff0000u);
  r.l.x = bf16_rne(l0) | (bf16_rne(l1) << 16);
  r.l.y = bf16_rne(l2) | (bf16_rne(l3) << 16);
  return r;
}

// ---------------------------------------------------------------------------
// Prep: pack W = [Wd(16x1024); Wo(240x1024)] (f32, row-major) into WMMA
// B-fragments in ws, split into hi (truncated bf16) and lo (RNE bf16 of
// residual). Fragment (nt,kt,h): 32 lanes x 16 bf16 (32B/lane).
// B layout (16-bit 32x16 KxN): lane<16 -> N=lane, K=kt*32+e (e=0..15);
//                              lane>=16 -> N=lane-16, K=kt*32+16+e.
// ---------------------------------------------------------------------------
__global__ __launch_bounds__(256) void wfrag_prep_kernel(
    const float* __restrict__ Wd, const float* __restrict__ Wo,
    unsigned char* __restrict__ ws) {
  int tid  = blockIdx.x * 256 + threadIdx.x;   // 32768 threads total
  int lane = tid & 31;
  int frag = tid >> 5;                         // (nt<<6) | (kt<<1) | h
  int h    = frag & 1;
  int kt   = (frag >> 1) & (NKT - 1);
  int nt   = frag >> 6;
  int n    = nt * 16 + (lane & 15);
  int kbase = kt * KT_SIZE + ((lane >> 4) << 4);   // +0 or +16

  const float* wrow = (n < 16) ? (Wd + (size_t)n * D_DIM)
                               : (Wo + (size_t)(n - 16) * D_DIM);
  const f32x4* src = (const f32x4*)(wrow + kbase);

  float v[16];
  #pragma unroll
  for (int q = 0; q < 4; ++q) {
    f32x4 tq = src[q];
    v[q * 4 + 0] = tq.x; v[q * 4 + 1] = tq.y;
    v[q * 4 + 2] = tq.z; v[q * 4 + 3] = tq.w;
  }

  unsigned pk[8];
  #pragma unroll
  for (int e = 0; e < 8; ++e) {
    float f0 = v[2 * e], f1 = v[2 * e + 1];
    unsigned b0, b1;
    if (h == 0) {                              // hi: truncate to bf16 (exact split)
      b0 = __float_as_uint(f0) >> 16;
      b1 = __float_as_uint(f1) >> 16;
    } else {                                   // lo: residual, RNE
      float h0 = __uint_as_float(__float_as_uint(f0) & 0xffff0000u);
      float h1 = __uint_as_float(__float_as_uint(f1) & 0xffff0000u);
      b0 = bf16_rne(f0 - h0);
      b1 = bf16_rne(f1 - h1);
    }
    pk[e] = b0 | (b1 << 16);
  }

  uint4* dst = (uint4*)(ws + ((size_t)frag * 32 + lane) * 32);
  uint4 q0; q0.x = pk[0]; q0.y = pk[1]; q0.z = pk[2]; q0.w = pk[3];
  uint4 q1; q1.x = pk[4]; q1.y = pk[5]; q1.z = pk[6]; q1.w = pk[7];
  dst[0] = q0;
  dst[1] = q1;
}

// ---------------------------------------------------------------------------
// Main: each block = 64 rows x 256 cols. K-chunk double-buffered pipeline:
//   iter kt: prefetch chunk kt+2 (global->regs), WMMA on LDS buf[kt&1],
//            convert staged chunk kt+1 -> LDS buf[(kt+1)&1], barrier.
// 8 waves each own 2 N-tiles x 4 M-tiles of f32 accumulators.
// ---------------------------------------------------------------------------
__global__ __launch_bounds__(256) void capmat_main_kernel(
    const float* __restrict__ x,
    const float* __restrict__ bd,
    const float* __restrict__ bo,
    const unsigned char* __restrict__ wfrag,
    float* __restrict__ out) {
  __shared__ __align__(16) unsigned char smem[SMEM_BYTES];

  const int t    = threadIdx.x;
  const int lane = t & 31;
  const int wave = t >> 5;
  const int nt0  = wave * 2;                 // this wave's two N-tiles

  // Loader mapping: chunk = 64 rows x 32 f32. Thread t handles float4 column
  // (t&7) of rows (t>>3) and (t>>3)+32 (8 threads x 128B contiguous per row).
  const int prow = t >> 3;                   // 0..31
  const int pcol = t & 7;                    // float4 col within chunk
  const f32x4* xg =
      (const f32x4*)(x + (size_t)blockIdx.x * ROWS_PER_BLOCK * D_DIM);

  // Per-lane LDS bases; all buffer/plane/mtile offsets are small immediates.
  unsigned char* st = smem + prow * CH_ROW_BYTES + pcol * 8;            // stores
  const unsigned char* pa = smem + (lane & 15) * CH_ROW_BYTES + (lane >> 4) * 32;

  f32x4 stage[2][2];   // [chunk parity][row half]

  auto load_chunk = [&](int kt, int parity) {
    int kc = (kt <= NKT - 1) ? kt : (NKT - 1);   // clamp (harmless reload)
    stage[parity][0] =
        __builtin_nontemporal_load(&xg[(size_t)prow * 256 + kc * 8 + pcol]);
    stage[parity][1] =
        __builtin_nontemporal_load(&xg[(size_t)(prow + 32) * 256 + kc * 8 + pcol]);
  };
  auto store_chunk = [&](int buf, int parity) {
    HiLo a = split4(stage[parity][0]);           // rows 0..31
    HiLo c = split4(stage[parity][1]);           // rows 32..63 (+2560B)
    *(uint2*)(st + buf * CH_BUF_BYTES + 0)                          = a.h;
    *(uint2*)(st + buf * CH_BUF_BYTES + CH_PLANE_BYTES)             = a.l;
    *(uint2*)(st + buf * CH_BUF_BYTES + 2560)                       = c.h;
    *(uint2*)(st + buf * CH_BUF_BYTES + CH_PLANE_BYTES + 2560)      = c.l;
  };

  // ---- Pipeline prologue: chunk0 -> buf0, chunk1 staged in regs.
  load_chunk(0, 0);
  load_chunk(1, 1);
  store_chunk(0, 0);
  __syncthreads();

  const uint4* wv = (const uint4*)wfrag;

  f32x8 acc[MT][2];
  #pragma unroll
  for (int m = 0; m < MT; ++m) { acc[m][0] = zero8(); acc[m][1] = zero8(); }

  // A layout (16-bit 16x32): lane<16 -> M=lane,  K = 0..15 (same order as B)
  //                          lane>=16 -> M=lane-16, K = 16..31
  #pragma unroll 2
  for (int kt = 0; kt < NKT; ++kt) {
    const int cur = kt & 1, nxt = cur ^ 1;

    // 1) Prefetch chunk kt+2 into the stage slot freed last iteration.
    load_chunk(kt + 2, cur);

    // 2) B fragments (hi/lo for both N-tiles) for this kt from L2.
    const int fiH0 = (nt0 * NKT + kt) * 2;
    const int fiH1 = ((nt0 + 1) * NKT + kt) * 2;
    const int iH0 = (fiH0 * 32 + lane) * 2;
    const int iL0 = ((fiH0 + 1) * 32 + lane) * 2;
    const int iH1 = (fiH1 * 32 + lane) * 2;
    const int iL1 = ((fiH1 + 1) * 32 + lane) * 2;
    bf16x16 Bh0 = make_frag(wv[iH0], wv[iH0 + 1]);
    bf16x16 Bl0 = make_frag(wv[iL0], wv[iL0 + 1]);
    bf16x16 Bh1 = make_frag(wv[iH1], wv[iH1 + 1]);
    bf16x16 Bl1 = make_frag(wv[iL1], wv[iL1 + 1]);

    // 3) WMMAs over 4 M-tiles reading LDS buf[cur] (offsets are immediates).
    #pragma unroll
    for (int m = 0; m < MT; ++m) {
      const int o = cur * CH_BUF_BYTES + m * (16 * CH_ROW_BYTES);
      bf16x16 Ah = make_frag(*(const uint4*)(pa + o),
                             *(const uint4*)(pa + o + 16));
      bf16x16 Al = make_frag(*(const uint4*)(pa + o + CH_PLANE_BYTES),
                             *(const uint4*)(pa + o + CH_PLANE_BYTES + 16));

      acc[m][0] = __builtin_amdgcn_wmma_f32_16x16x32_bf16(false, Ah, false, Bl0,
                      (short)0, acc[m][0], false, false);
      acc[m][0] = __builtin_amdgcn_wmma_f32_16x16x32_bf16(false, Al, false, Bh0,
                      (short)0, acc[m][0], false, false);
      acc[m][0] = __builtin_amdgcn_wmma_f32_16x16x32_bf16(false, Ah, false, Bh0,
                      (short)0, acc[m][0], false, false);

      acc[m][1] = __builtin_amdgcn_wmma_f32_16x16x32_bf16(false, Ah, false, Bl1,
                      (short)0, acc[m][1], false, false);
      acc[m][1] = __builtin_amdgcn_wmma_f32_16x16x32_bf16(false, Al, false, Bh1,
                      (short)0, acc[m][1], false, false);
      acc[m][1] = __builtin_amdgcn_wmma_f32_16x16x32_bf16(false, Ah, false, Bh1,
                      (short)0, acc[m][1], false, false);
    }

    // 4) Convert staged chunk kt+1 into the other LDS buffer.
    store_chunk(nxt, nxt);
    __syncthreads();
  }

  // ---- Epilogue: bias + activation + scatter into C[b, i, j].
  // C/D layout: VGPR r, lane l -> row M = r + (l<16?0:8), col N = l&15.
  #pragma unroll
  for (int j = 0; j < 2; ++j) {
    const int n = (nt0 + j) * 16 + (lane & 15);
    float bias;
    int oi, oj;
    const bool diag = (n < 16);               // wave-uniform (tile 0 only)
    if (diag) {
      bias = bd[n]; oi = n; oj = n;
    } else {
      int r2 = n - 16;
      oi = r2 / 15;
      int jj = r2 % 15;
      oj = jj + (jj >= oi ? 1 : 0);
      bias = bo[r2];
    }
    const int rowadd = (lane >> 4) << 3;      // +0 or +8
    #pragma unroll
    for (int m = 0; m < MT; ++m) {
      #pragma unroll
      for (int r = 0; r < 8; ++r) {
        float y = acc[m][j][r] + bias;
        float val;
        if (diag) {
          val = 1.0f / (1.0f + __expf(-y));                       // sigmoid
        } else {
          float sp = fmaxf(y, 0.0f) + log1pf(__expf(-fabsf(y)));  // softplus
          val = -sp;
        }
        const int row = m * 16 + r + rowadd;
        const size_t b = (size_t)blockIdx.x * ROWS_PER_BLOCK + row;
        __builtin_nontemporal_store(val, &out[b * 256 + (size_t)(oi * 16 + oj)]);
      }
    }
  }
}

// ---------------------------------------------------------------------------
extern "C" void kernel_launch(void* const* d_in, const int* in_sizes, int n_in,
                              void* d_out, int out_size, void* d_ws, size_t ws_size,
                              hipStream_t stream) {
  (void)in_sizes; (void)n_in; (void)out_size; (void)ws_size;
  const float* x  = (const float*)d_in[0];   // (65536, 1024)
  const float* Wd = (const float*)d_in[1];   // (16, 1024)
  const float* bd = (const float*)d_in[2];   // (16,)
  const float* Wo = (const float*)d_in[3];   // (240, 1024)
  const float* bo = (const float*)d_in[4];   // (240,)
  float* out = (float*)d_out;                // (65536, 16, 16) f32
  unsigned char* ws = (unsigned char*)d_ws;  // needs WS_BYTES_NEEDED = 1 MiB

  // 1) Pack weights into bf16 hi/lo WMMA B-fragments (1 MiB in ws, ~free).
  wfrag_prep_kernel<<<128, 256, 0, stream>>>(Wd, Wo, ws);
  // 2) Fused GEMM + activation + scatter. 1024 blocks x 256 threads, 20KB LDS.
  capmat_main_kernel<<<65536 / ROWS_PER_BLOCK, 256, 0, stream>>>(
      x, bd, bo, ws, out);
}
